// SmoothMoNet_57861799412198
// MI455X (gfx1250) — compile-verified, matrix operand order checked by reading, not run on hardware
//
#include <hip/hip_runtime.h>
#include <hip/hip_bf16.h>

typedef __attribute__((ext_vector_type(16))) _Float16 v16h;
typedef __attribute__((ext_vector_type(8)))  float    v8f;

#define HID 64
#define OUTD 64
#define KKER 3
#define NCLS 6
#define NLAY 4
#define COLS 192   // K*OUTD

// ---------------------------------------------------------------- utility
__global__ void zero_k(float* __restrict__ p, size_t n) {
  size_t t = (size_t)blockIdx.x * blockDim.x + threadIdx.x;
  if (t < n) p[t] = 0.f;
}

// ---------------------------------------------------------------- degrees
__global__ void indeg_k(const int* __restrict__ vs, float* __restrict__ deg, int E) {
  int e = blockIdx.x * blockDim.x + threadIdx.x;
  if (e < E) atomicAdd(&deg[vs[e]], 1.0f);
}

__global__ void invdeg_k(float* __restrict__ deg, int N) {
  int n = blockIdx.x * blockDim.x + threadIdx.x;
  if (n < N) deg[n] = rsqrtf(deg[n] + 1.0f);   // in-place: deg -> inv
}

// ---------------------------------------------------------------- embedding
__global__ void embed_k(const int* __restrict__ h, const float* __restrict__ emb,
                        float* __restrict__ x, int N) {
  int t = blockIdx.x * blockDim.x + threadIdx.x;
  if (t >= N * HID) return;
  int n = t >> 6, c = t & 63;
  x[t] = emb[h[n] * HID + c];
}

// ------------------------------------------------ per-edge Gaussian weights
struct GmmPP {
  const float* pp_w[NLAY];
  const float* pp_b[NLAY];
  const float* mu[NLAY];
  const float* isig[NLAY];
};

__global__ void edge_w_k(const int* __restrict__ us, const int* __restrict__ vs,
                         const float* __restrict__ inv, GmmPP pp,
                         float* __restrict__ wAll, int E) {
  int e = blockIdx.x * blockDim.x + threadIdx.x;
  if (e >= E) return;
  float pu = inv[us[e]], pv = inv[vs[e]];
#pragma unroll
  for (int l = 0; l < NLAY; ++l) {
    const float* W = pp.pp_w[l];    // (2,2) row-major
    const float* B = pp.pp_b[l];
    float pz0 = tanhf(pu * W[0] + pv * W[2] + B[0]);
    float pz1 = tanhf(pu * W[1] + pv * W[3] + B[1]);
    const float* mu = pp.mu[l];     // (3,2)
    const float* sg = pp.isig[l];   // (3,2)
#pragma unroll
    for (int k = 0; k < KKER; ++k) {
      float d0 = (pz0 - mu[2 * k])     * sg[2 * k];
      float d1 = (pz1 - mu[2 * k + 1]) * sg[2 * k + 1];
      wAll[(size_t)l * KKER * E + (size_t)e * KKER + k] =
          __expf(-0.5f * (d0 * d0 + d1 * d1));
    }
  }
}

// ---------------------------------------------------------------- WMMA GEMM
// hk(N x 192) = X(N x 64) @ W(64 x 192), f16 inputs / f32 accumulate.
// One wave per 16-row tile; A fragment stays in registers across all 12
// column tiles -> 24 v_wmma per wave, 12x less X traffic.
__global__ __launch_bounds__(256) void gemm_hk_k(const float* __restrict__ X,
                                                 const float* __restrict__ W,
                                                 float* __restrict__ Hk,
                                                 int nRowTiles) {
  int wid  = (blockIdx.x * blockDim.x + threadIdx.x) >> 5;  // wave id = row tile
  int lane = threadIdx.x & 31;
  if (wid >= nRowTiles) return;          // whole-wave exit: EXEC stays all-1s
  int row0 = wid * 16;

  int m    = lane & 15;
  int kofs = (lane < 16) ? 0 : 8;        // ISA 16-bit A/B lane K-offset
  const float* xr = X + (size_t)(row0 + m) * HID;
  __builtin_prefetch(X + (size_t)(row0 + 16 + m) * HID, 0, 1);  // next row tile

  // A fragments for K=0..31 and K=32..63, kept in VGPRs for all col tiles
  v16h a0, a1;
#pragma unroll
  for (int e = 0; e < 8; ++e) {
    a0[e]     = (_Float16)xr[kofs + e];            // K = kofs..kofs+7
    a0[e + 8] = (_Float16)xr[kofs + 16 + e];       // K = kofs+16..kofs+23
    a1[e]     = (_Float16)xr[32 + kofs + e];
    a1[e + 8] = (_Float16)xr[32 + kofs + 16 + e];
  }

  int rbase = row0 + ((lane < 16) ? 0 : 8);        // C/D layout row base
#pragma unroll 4
  for (int ct = 0; ct < COLS / 16; ++ct) {
    int col = ct * 16 + m;
    v16h b0, b1;
#pragma unroll
    for (int e = 0; e < 8; ++e) {
      b0[e]     = (_Float16)W[(kofs + e) * COLS + col];
      b0[e + 8] = (_Float16)W[(kofs + 16 + e) * COLS + col];
      b1[e]     = (_Float16)W[(32 + kofs + e) * COLS + col];
      b1[e + 8] = (_Float16)W[(32 + kofs + 16 + e) * COLS + col];
    }
    v8f c = {};
    c = __builtin_amdgcn_wmma_f32_16x16x32_f16(false, a0, false, b0,
                                               (short)0, c, false, false);
    c = __builtin_amdgcn_wmma_f32_16x16x32_f16(false, a1, false, b1,
                                               (short)0, c, false, false);
    float* out = Hk + (size_t)rbase * COLS + col;
#pragma unroll
    for (int r = 0; r < 8; ++r) out[(size_t)r * COLS] = c[r];
  }
}

// ------------------------------------------------ edge gather + scatter-add
// 64 lanes (2 waves) per edge: lane j handles feature j. hk/agg are L2-resident.
__global__ __launch_bounds__(256) void edge_agg_k(const int* __restrict__ us,
                                                  const int* __restrict__ vs,
                                                  const float* __restrict__ Hk,
                                                  const float* __restrict__ wL,
                                                  float* __restrict__ agg, int E) {
  int tid = threadIdx.x;
  int e   = blockIdx.x * 4 + (tid >> 6);
  int j   = tid & 63;
  if (e >= E) return;
  int u = us[e], v = vs[e];
  float w0 = wL[(size_t)e * KKER + 0];
  float w1 = wL[(size_t)e * KKER + 1];
  float w2 = wL[(size_t)e * KKER + 2];
  const float* hr = Hk + (size_t)u * COLS;
  float msg = w0 * hr[j] + w1 * hr[HID + j] + w2 * hr[2 * HID + j];
  atomicAdd(&agg[(size_t)v * HID + j], msg);
}

// ---------------------------------------------------------------- batchnorm
__global__ void bn_stats_k(const float* __restrict__ agg,
                           const float* __restrict__ snorm,
                           float* __restrict__ stats, int N) {
  int tid = blockIdx.x * blockDim.x + threadIdx.x;
  int c = tid & 63;
  int stride = (gridDim.x * blockDim.x) >> 6;
  float s = 0.f, s2 = 0.f;
  for (int n = tid >> 6; n < N; n += stride) {
    float v = agg[(size_t)n * HID + c] * snorm[n];
    s += v; s2 += v * v;
  }
  atomicAdd(&stats[c], s);
  atomicAdd(&stats[64 + c], s2);
}

__global__ void bn_fin_k(const float* __restrict__ stats,
                         const float* __restrict__ g, const float* __restrict__ b,
                         float* __restrict__ bnp, int N) {
  int c = threadIdx.x;
  if (c >= HID) return;
  float mean = stats[c] / (float)N;
  float var  = stats[64 + c] / (float)N - mean * mean;
  float sc   = rsqrtf(var + 1e-5f) * g[c];
  bnp[c]      = sc;
  bnp[64 + c] = b[c] - mean * sc;
}

__global__ void bn_apply_k(float* __restrict__ x, const float* __restrict__ agg,
                           const float* __restrict__ snorm,
                           const float* __restrict__ bnp, int N) {
  int t = blockIdx.x * blockDim.x + threadIdx.x;
  if (t >= N * HID) return;
  int n = t >> 6, c = t & 63;
  float hh = agg[t] * snorm[n];
  float v  = hh * bnp[c] + bnp[64 + c];
  x[t] += fmaxf(v, 0.f);
}

// ---------------------------------------------------------------- readout p
__global__ __launch_bounds__(128) void readout_p_k(const float* __restrict__ x,
    const float* __restrict__ W0, const float* __restrict__ B0,
    const float* __restrict__ W1, const float* __restrict__ B1,
    const float* __restrict__ W2, const float* __restrict__ B2,
    float* __restrict__ out, int N) {
  __shared__ float sW0[64 * 32], sW1[32 * 16], sW2[16 * 6];
  __shared__ float sB0[32], sB1[16], sB2[6];
  __shared__ float sX[128 * HID];
  int tid = threadIdx.x;
  for (int i = tid; i < 64 * 32; i += 128) sW0[i] = W0[i];
  for (int i = tid; i < 32 * 16; i += 128) sW1[i] = W1[i];
  for (int i = tid; i < 16 * 6;  i += 128) sW2[i] = W2[i];
  if (tid < 32) sB0[tid] = B0[tid];
  if (tid < 16) sB1[tid] = B1[tid];
  if (tid < 6)  sB2[tid] = B2[tid];
  int base = blockIdx.x * 128;
  int cnt  = N - base; if (cnt > 128) cnt = 128;
  for (int i = tid; i < cnt * HID; i += 128) sX[i] = x[(size_t)base * HID + i];
  __syncthreads();
  if (tid >= cnt) return;
  const float* xr = sX + tid * HID;
  float h1[32];
#pragma unroll 4
  for (int j = 0; j < 32; ++j) {
    float a = sB0[j];
    for (int i = 0; i < 64; ++i) a += xr[i] * sW0[i * 32 + j];
    h1[j] = fmaxf(a, 0.f);
  }
  float h2[16];
#pragma unroll 4
  for (int j = 0; j < 16; ++j) {
    float a = sB1[j];
    for (int i = 0; i < 32; ++i) a += h1[i] * sW1[i * 16 + j];
    h2[j] = fmaxf(a, 0.f);
  }
  float* po = out + (size_t)(base + tid) * NCLS;
#pragma unroll
  for (int j = 0; j < NCLS; ++j) {
    float a = sB2[j];
    for (int i = 0; i < 16; ++i) a += h2[i] * sW2[i * 6 + j];
    po[j] = a;
  }
}

// -------------------------------------------------------------- readout g_hat
#define RD (HID + NCLS)   // 70
__global__ __launch_bounds__(128) void readout_g_k(const float* __restrict__ x,
    const float* __restrict__ label,
    const float* __restrict__ Wr0, const float* __restrict__ Br0,
    const float* __restrict__ Wr1, const float* __restrict__ Br1,
    const float* __restrict__ Wf,  const float* __restrict__ Bf,
    float* __restrict__ out, int N) {
  __shared__ float sW0[RD * RD], sW1[RD * RD], sB0[RD], sB1[RD], sWf[RD], sBf;
  int tid = threadIdx.x;
  for (int i = tid; i < RD * RD; i += 128) { sW0[i] = Wr0[i]; sW1[i] = Wr1[i]; }
  if (tid < RD) { sB0[tid] = Br0[tid]; sB1[tid] = Br1[tid]; sWf[tid] = Wf[tid]; }
  if (tid == 0) sBf = Bf[0];
  __syncthreads();
  int n = blockIdx.x * 128 + tid;
  if (n >= N) return;
  float y[RD], t[RD];
  for (int i = 0; i < HID;  ++i) y[i]       = x[(size_t)n * HID + i];
  for (int i = 0; i < NCLS; ++i) y[HID + i] = label[(size_t)n * NCLS + i];
  for (int j = 0; j < RD; ++j) {
    float a = sB0[j];
    for (int i = 0; i < RD; ++i) a += y[i] * sW0[i * RD + j];
    t[j] = fmaxf(a, 0.f) + y[j];
  }
  for (int j = 0; j < RD; ++j) {
    float a = sB1[j];
    for (int i = 0; i < RD; ++i) a += t[i] * sW1[i * RD + j];
    y[j] = fmaxf(a, 0.f) + t[j];
  }
  float s = sBf;
  for (int i = 0; i < RD; ++i) s += y[i] * sWf[i];
  float w = 1.f / (1.f + __expf(-s));
  w = fminf(fmaxf(w, 0.f), 0.5f);
  const float* lab = label + (size_t)n * NCLS;
  float* go = out + (size_t)N * NCLS + (size_t)n * NCLS;
#pragma unroll
  for (int k = 0; k < NCLS; ++k) go[k] = (1.f - w) * lab[k] + w * (1.f / 6.f);
}

// ================================================================ launcher
extern "C" void kernel_launch(void* const* d_in, const int* in_sizes, int n_in,
                              void* d_out, int out_size, void* d_ws, size_t ws_size,
                              hipStream_t stream) {
  const int N = in_sizes[0];
  const int E = in_sizes[1];

  const int*   h      = (const int*)d_in[0];
  const int*   us     = (const int*)d_in[1];
  const int*   vs     = (const int*)d_in[2];
  const float* snorm  = (const float*)d_in[3];
  const float* label  = (const float*)d_in[4];
  const float* emb    = (const float*)d_in[5];
  // per-layer (pytree leaves, keys sorted): bn_b, bn_g, fc_w, inv_sigma, mu, pp_b, pp_w
  const float *bn_b[NLAY], *bn_g[NLAY], *fc_w[NLAY];
  GmmPP pp;
  for (int l = 0; l < NLAY; ++l) {
    int b = 6 + 7 * l;
    bn_b[l]     = (const float*)d_in[b + 0];
    bn_g[l]     = (const float*)d_in[b + 1];
    fc_w[l]     = (const float*)d_in[b + 2];
    pp.isig[l]  = (const float*)d_in[b + 3];
    pp.mu[l]    = (const float*)d_in[b + 4];
    pp.pp_b[l]  = (const float*)d_in[b + 5];
    pp.pp_w[l]  = (const float*)d_in[b + 6];
  }
  const float* mlpB0 = (const float*)d_in[34];
  const float* mlpW0 = (const float*)d_in[35];
  const float* mlpB1 = (const float*)d_in[36];
  const float* mlpW1 = (const float*)d_in[37];
  const float* mlpB2 = (const float*)d_in[38];
  const float* mlpW2 = (const float*)d_in[39];
  const float* wfB   = (const float*)d_in[40];
  const float* wfW   = (const float*)d_in[41];
  const float* wrB0  = (const float*)d_in[42];
  const float* wrW0  = (const float*)d_in[43];
  const float* wrB1  = (const float*)d_in[44];
  const float* wrW1  = (const float*)d_in[45];

  float* out = (float*)d_out;

  // ---- workspace layout (floats)
  float* ws = (float*)d_ws;
  size_t o = 0;
  float* inv   = ws + o; o += ((size_t)N + 63) & ~63ull;         // N
  float* x     = ws + o; o += (size_t)N * HID;                   // N*64
  float* hk    = ws + o; o += (size_t)N * COLS;                  // N*192
  float* agg   = ws + o; o += (size_t)N * HID;                   // N*64
  float* wAll  = ws + o; o += (size_t)NLAY * KKER * E;           // 4*3*E
  float* stats = ws + o; o += 128;
  float* bnp   = ws + o; o += 128;
  (void)ws_size; (void)n_in; (void)out_size;

  const int T = 256;
  // degrees -> inv
  zero_k<<<(N + T - 1) / T, T, 0, stream>>>(inv, (size_t)N);
  indeg_k<<<(E + T - 1) / T, T, 0, stream>>>(vs, inv, E);
  invdeg_k<<<(N + T - 1) / T, T, 0, stream>>>(inv, N);
  // Gaussian edge weights, all layers
  edge_w_k<<<(E + T - 1) / T, T, 0, stream>>>(us, vs, inv, pp, wAll, E);
  // embedding
  embed_k<<<((size_t)N * HID + T - 1) / T, T, 0, stream>>>(h, emb, x, N);

  const int nRowTiles  = (N + 15) / 16;
  const int gemmBlocks = (nRowTiles * 32 + T - 1) / T;   // one wave per row tile

  for (int l = 0; l < NLAY; ++l) {
    gemm_hk_k<<<gemmBlocks, T, 0, stream>>>(x, fc_w[l], hk, nRowTiles);
    zero_k<<<((size_t)N * HID + T - 1) / T, T, 0, stream>>>(agg, (size_t)N * HID);
    zero_k<<<1, 128, 0, stream>>>(stats, 128);
    edge_agg_k<<<(E + 3) / 4, T, 0, stream>>>(us, vs, hk,
                                              wAll + (size_t)l * KKER * E, agg, E);
    bn_stats_k<<<256, T, 0, stream>>>(agg, snorm, stats, N);
    bn_fin_k<<<1, HID, 0, stream>>>(stats, bn_g[l], bn_b[l], bnp, N);
    bn_apply_k<<<((size_t)N * HID + T - 1) / T, T, 0, stream>>>(x, agg, snorm, bnp, N);
  }

  readout_p_k<<<(N + 127) / 128, 128, 0, stream>>>(x, mlpW0, mlpB0, mlpW1, mlpB1,
                                                   mlpW2, mlpB2, out, N);
  readout_g_k<<<(N + 127) / 128, 128, 0, stream>>>(x, label, wrW0, wrB0, wrW1, wrB1,
                                                   wfW, wfB, out, N);
}
